// BiGCN_dgl_17119739641944
// MI455X (gfx1250) — matrix operand order repaired and checked by reading, not compiled, not run on hardware
//
#include <hip/hip_runtime.h>
#include <hip/hip_bf16.h>

// ---------------------------------------------------------------------------
// BiGCN (3-relation heterogeneous GCN, 2 layers) + edge/pair link scoring.
// Dense GEMMs: CDNA5 WMMA (bf16 in, f32 accum), 16x64 strip per wave.
// SpMM: atomic-free via device-built CSR (dst-grouped) + gather-reduce.
// ---------------------------------------------------------------------------

typedef __attribute__((ext_vector_type(16))) __bf16 v16bf;
typedef __attribute__((ext_vector_type(8)))  float  v8f;

struct U32x8 { uint4 a, b; };
union FragCast { U32x8 q; v16bf v; };

__device__ __forceinline__ unsigned short f2bf(float f) {
    unsigned int u = __float_as_uint(f);
    unsigned int r = u + 0x7FFFu + ((u >> 16) & 1u);   // round-to-nearest-even
    return (unsigned short)(r >> 16);
}

__device__ __forceinline__ int atomAddI(int* p, int v) {
    return __hip_atomic_fetch_add(p, v, __ATOMIC_RELAXED, __HIP_MEMORY_SCOPE_AGENT);
}

// ------------------------------ utility kernels ----------------------------

__global__ void k_zero_f32(float* __restrict__ p, long long n) {
    long long i = (long long)blockIdx.x * blockDim.x + threadIdx.x;
    if (i < n) p[i] = 0.0f;
}

__global__ void k_zero_i32(int* __restrict__ p, long long n) {
    long long i = (long long)blockIdx.x * blockDim.x + threadIdx.x;
    if (i < n) p[i] = 0;
}

__global__ void k_copy_i32(const int* __restrict__ s, int* __restrict__ d, long long n) {
    long long i = (long long)blockIdx.x * blockDim.x + threadIdx.x;
    if (i < n) d[i] = s[i];
}

// int degree histograms for all relations
__global__ void k_count(const int* __restrict__ src, const int* __restrict__ dst,
                        int* __restrict__ cOut, int* __restrict__ cIn,
                        int total, int E, int N) {
    int t = blockIdx.x * blockDim.x + threadIdx.x;
    if (t >= total) return;
    int r = t / E;
    atomAddI(&cOut[(size_t)r * N + src[t]], 1);
    atomAddI(&cIn [(size_t)r * N + dst[t]], 1);
}

__global__ void k_scales(const int* __restrict__ cnt, float* __restrict__ s, long long n) {
    long long i = (long long)blockIdx.x * blockDim.x + threadIdx.x;
    if (i < n) s[i] = rsqrtf((float)max(cnt[i], 1));
}

// --------------------- exclusive scan (3 kernels, CHUNK=1024) --------------

__global__ void k_scan1(const int* __restrict__ cnt, int* __restrict__ excl,
                        int* __restrict__ chunkTot, int tot) {
    __shared__ int sd[256];
    int tid = threadIdx.x;
    int base = blockIdx.x * 1024 + tid * 4;
    int v0 = 0, v1 = 0, v2 = 0, v3 = 0;
    if (base + 0 < tot) v0 = cnt[base + 0];
    if (base + 1 < tot) v1 = cnt[base + 1];
    if (base + 2 < tot) v2 = cnt[base + 2];
    if (base + 3 < tot) v3 = cnt[base + 3];
    int s4 = v0 + v1 + v2 + v3;
    sd[tid] = s4;
    __syncthreads();
    for (int off = 1; off < 256; off <<= 1) {
        int v = (tid >= off) ? sd[tid - off] : 0;
        __syncthreads();
        sd[tid] += v;
        __syncthreads();
    }
    int run = sd[tid] - s4;                 // exclusive offset of this thread
    if (base + 0 < tot) excl[base + 0] = run;            run += v0;
    if (base + 1 < tot) excl[base + 1] = run;            run += v1;
    if (base + 2 < tot) excl[base + 2] = run;            run += v2;
    if (base + 3 < tot) excl[base + 3] = run;
    if (tid == 255) chunkTot[blockIdx.x] = sd[255];
}

__global__ void k_scan2(int* __restrict__ chunkTot, int nChunks) {
    __shared__ int sd[512];
    int tid = threadIdx.x;
    int v = (tid < nChunks) ? chunkTot[tid] : 0;
    sd[tid] = v;
    __syncthreads();
    for (int off = 1; off < 512; off <<= 1) {
        int u = (tid >= off) ? sd[tid - off] : 0;
        __syncthreads();
        sd[tid] += u;
        __syncthreads();
    }
    if (tid < nChunks) chunkTot[tid] = sd[tid] - v;      // exclusive
}

__global__ void k_scan3(int* __restrict__ excl, const int* __restrict__ chunkExcl, int tot) {
    int i = blockIdx.x * blockDim.x + threadIdx.x;
    if (i < tot) excl[i] += chunkExcl[i >> 10];
}

// bucket-fill: eidx grouped by CSR row (r*N + dst)
__global__ void k_fill(const int* __restrict__ dst, int* __restrict__ cursor,
                       int* __restrict__ eidx, int total, int E, int N) {
    int t = blockIdx.x * blockDim.x + threadIdx.x;
    if (t >= total) return;
    int row = (t / E) * N + dst[t];
    int pos = atomAddI(&cursor[row], 1);
    eidx[pos] = t;
}

// ----------------------------- bf16 staging --------------------------------

// out[n,k] = bf16( X[n,k] * rowScale[n] )
__global__ void k_scale_to_bf16(const float* __restrict__ X,
                                const float* __restrict__ rowScale,
                                unsigned short* __restrict__ out,
                                int ncols, long long total) {
    long long t = (long long)blockIdx.x * blockDim.x + threadIdx.x;
    if (t >= total) return;
    long long n = t / ncols;
    out[t] = f2bf(X[t] * rowScale[n]);
}

// W [R][KIN][KOUT] f32 row-major  ->  WT [R][KOUT][KIN] bf16
__global__ void k_transpose_bf16(const float* __restrict__ W,
                                 unsigned short* __restrict__ WT,
                                 int KIN, int KOUT, int total) {
    int t = blockIdx.x * blockDim.x + threadIdx.x;
    if (t >= total) return;
    int per = KIN * KOUT;
    int r = t / per, rem = t % per;
    int n = rem / KIN, k = rem % KIN;
    WT[t] = f2bf(W[(size_t)r * per + (size_t)k * KOUT + n]);
}

// ------------------------------ WMMA GEMM ----------------------------------
// C[M, NC] = A[M, K](bf16) @ BT[NC, K](bf16, pre-transposed), f32 accumulate.
// Each wave computes a 16x(16*TPW) strip: the A fragment is loaded once per
// 32-K chunk and reused across TPW WMMAs (4x A-traffic reduction vs 1 tile).
// A-frag (16-bit A 16x32):  lane half h: K = i + (i>=8?8:0) + 8h  -> 2x16B loads
// B-frag (16-bit B 32x16):  lane half h: K = i + (i>=8?8:0) + 8h  -> 2x16B loads
template <int K, int NC, int WAVES>
__global__ void k_gemm_wmma(const unsigned short* __restrict__ A,
                            const unsigned short* __restrict__ BT,
                            float* __restrict__ C) {
    constexpr int TPW = (NC / 16) / WAVES;   // n-tiles per wave
    const int lane = threadIdx.x & 31;
    const int wave = threadIdx.x >> 5;
    const int mTile = blockIdx.x;
    const int nTile0 = wave * TPW;
    const int half = lane >> 4;
    const int mr = lane & 15;

    const uint4* Arow = reinterpret_cast<const uint4*>(A + (size_t)(mTile * 16 + mr) * K);
    const uint4* Bcol[TPW];
#pragma unroll
    for (int tt = 0; tt < TPW; ++tt)
        Bcol[tt] = reinterpret_cast<const uint4*>(BT + (size_t)((nTile0 + tt) * 16 + mr) * K);

    v8f acc[TPW];
#pragma unroll
    for (int tt = 0; tt < TPW; ++tt) acc[tt] = (v8f){};

#pragma unroll
    for (int kb = 0; kb < K / 32; ++kb) {
        FragCast fa;
        fa.q.a = Arow[kb * 4 + half];
        fa.q.b = Arow[kb * 4 + half + 2];
#pragma unroll
        for (int tt = 0; tt < TPW; ++tt) {
            FragCast fb;
            fb.q.a = Bcol[tt][kb * 4 + half];
            fb.q.b = Bcol[tt][kb * 4 + half + 2];
            acc[tt] = __builtin_amdgcn_wmma_f32_16x16x32_bf16(
                false, fa.v, false, fb.v, (short)0, acc[tt], false, false);
        }
    }

    // C fragment: VGPR j -> row M = j + 8*half, col N = mr
#pragma unroll
    for (int tt = 0; tt < TPW; ++tt) {
#pragma unroll
        for (int j = 0; j < 8; ++j) {
            C[(size_t)(mTile * 16 + j + 8 * half) * NC + (nTile0 + tt) * 16 + mr] = acc[tt][j];
        }
    }
}

// ------------------ CSR gather-reduce (atomic-free SpMM) -------------------
// acc[n, :] += dstScale[n] * sum_{e in CSR row} h[src[e], :]
template <int NC>
__global__ void k_gather(const float* __restrict__ h,
                         const int* __restrict__ esrcFull,
                         const int* __restrict__ eidx,
                         const int* __restrict__ offs,
                         const int* __restrict__ cnt,
                         const float* __restrict__ dscale,
                         float* __restrict__ acc,
                         int N, int rowBase) {
    const int CPT = NC / 4;
    int t = blockIdx.x * blockDim.x + threadIdx.x;
    if (t >= N * CPT) return;
    int n = t / CPT, c = t % CPT;
    int row = rowBase + n;
    int beg = offs[row], num = cnt[row];
    const float4* h4 = reinterpret_cast<const float4*>(h);
    float4 s = make_float4(0.f, 0.f, 0.f, 0.f);
    for (int j = 0; j < num; ++j) {
        int e = eidx[beg + j];
        float4 v = h4[(size_t)esrcFull[e] * CPT + c];
        s.x += v.x; s.y += v.y; s.z += v.z; s.w += v.w;
    }
    float sc = dscale[n];
    float4* a4 = reinterpret_cast<float4*>(acc);
    float4 a = a4[(size_t)n * CPT + c];
    a.x += s.x * sc; a.y += s.y * sc; a.z += s.z * sc; a.w += s.w * sc;
    a4[(size_t)n * CPT + c] = a;
}

// mean over relations + mean bias (+ optional relu)
template <int NC, bool RELU>
__global__ void k_finalize(float* __restrict__ acc, const float* __restrict__ b,
                           long long total) {
    long long t = (long long)blockIdx.x * blockDim.x + threadIdx.x;
    if (t >= total) return;
    int k = (int)(t % NC);
    float mb = (b[k] + b[NC + k] + b[2 * NC + k]) * (1.0f / 3.0f);
    float v = acc[t] * (1.0f / 3.0f) + mb;
    acc[t] = RELU ? fmaxf(v, 0.0f) : v;
}

// ------------------------- final gather + dot + sigmoid --------------------
__global__ void k_score(const float* __restrict__ h,
                        const int* __restrict__ esrc, const int* __restrict__ edst,
                        const int* __restrict__ pairs,
                        const float* __restrict__ linW, const float* __restrict__ linb,
                        float* __restrict__ out, int nEdgeRows, int nTotal) {
    int lane = threadIdx.x & 31;
    long long row = ((long long)blockIdx.x * blockDim.x + threadIdx.x) >> 5;
    if (row >= nTotal) return;
    int a, b;
    if (row < nEdgeRows) { a = esrc[row]; b = edst[row]; }
    else {
        long long p = row - nEdgeRows;
        a = pairs[2 * p];
        b = pairs[2 * p + 1];
    }
    float4 ha = reinterpret_cast<const float4*>(h + (size_t)a * 128)[lane];
    float4 hb = reinterpret_cast<const float4*>(h + (size_t)b * 128)[lane];
    float4 wa = reinterpret_cast<const float4*>(linW)[lane];
    float4 wb = reinterpret_cast<const float4*>(linW + 128)[lane];
    float part = fmaxf(ha.x, 0.f) * wa.x + fmaxf(ha.y, 0.f) * wa.y +
                 fmaxf(ha.z, 0.f) * wa.z + fmaxf(ha.w, 0.f) * wa.w +
                 fmaxf(hb.x, 0.f) * wb.x + fmaxf(hb.y, 0.f) * wb.y +
                 fmaxf(hb.z, 0.f) * wb.z + fmaxf(hb.w, 0.f) * wb.w;
#pragma unroll
    for (int off = 16; off > 0; off >>= 1) part += __shfl_xor(part, off, 32);
    if (lane == 0) out[row] = 1.0f / (1.0f + __expf(-(part + linb[0])));
}

// ---------------------------------------------------------------------------

extern "C" void kernel_launch(void* const* d_in, const int* in_sizes, int n_in,
                              void* d_out, int out_size, void* d_ws, size_t ws_size,
                              hipStream_t stream) {
    const float* x     = (const float*)d_in[0];   // [N,128]
    const int*   esrc  = (const int*)  d_in[1];   // [3,E]
    const int*   edst  = (const int*)  d_in[2];   // [3,E]
    const int*   pairs = (const int*)  d_in[4];   // [P,2]
    const float* W1    = (const float*)d_in[5];   // [3,128,256]
    const float* b1    = (const float*)d_in[6];   // [3,256]
    const float* W2    = (const float*)d_in[7];   // [3,256,128]
    const float* b2    = (const float*)d_in[8];   // [3,128]
    const float* linW  = (const float*)d_in[9];   // [256]
    const float* linb  = (const float*)d_in[10];  // [1]
    float* out = (float*)d_out;

    const int N = in_sizes[0] / 128;              // 100000
    const int E = in_sizes[1] / 3;                // 400000
    const int TOT = 3 * N;                        // CSR rows
    const int nEdgeRows = 3 * E;
    const int nTotal = out_size;                  // 1400000
    const int nChunks = (TOT + 1023) / 1024;      // 293

    // ---- workspace carve-out (256B aligned regions) ----
    char* ws = (char*)d_ws;
    size_t off = 0;
    auto carve = [&](size_t bytes) -> char* {
        char* p = ws + off;
        off = (off + bytes + 255) & ~(size_t)255;
        return p;
    };
    int*   cntOut  = (int*)carve((size_t)TOT * 4);
    int*   cntIn   = (int*)carve((size_t)TOT * 4);
    float* sOut    = (float*)carve((size_t)TOT * 4);
    float* sIn     = (float*)carve((size_t)TOT * 4);
    int*   offs    = (int*)carve((size_t)TOT * 4);
    int*   cursor  = (int*)carve((size_t)TOT * 4);
    int*   chunkT  = (int*)carve(512 * 4);
    int*   eidx    = (int*)carve((size_t)nEdgeRows * 4);
    float* acc1    = (float*)carve((size_t)N * 256 * 4);
    float* tmp     = (float*)carve((size_t)N * 256 * 4);
    float* acc2    = (float*)carve((size_t)N * 128 * 4);
    unsigned short* Abf = (unsigned short*)carve((size_t)N * 256 * 2);
    unsigned short* W1T = (unsigned short*)carve((size_t)3 * 256 * 128 * 2);
    unsigned short* W2T = (unsigned short*)carve((size_t)3 * 128 * 256 * 2);

    const int BT = 256;
    auto blocks = [&](long long n) { return (unsigned)((n + BT - 1) / BT); };

    // ---- init + weights ----
    k_zero_i32<<<blocks(TOT), BT, 0, stream>>>(cntOut, TOT);
    k_zero_i32<<<blocks(TOT), BT, 0, stream>>>(cntIn, TOT);
    k_zero_f32<<<blocks((long long)N * 256), BT, 0, stream>>>(acc1, (long long)N * 256);
    k_zero_f32<<<blocks((long long)N * 128), BT, 0, stream>>>(acc2, (long long)N * 128);
    k_transpose_bf16<<<blocks(3 * 128 * 256), BT, 0, stream>>>(W1, W1T, 128, 256, 3 * 128 * 256);
    k_transpose_bf16<<<blocks(3 * 256 * 128), BT, 0, stream>>>(W2, W2T, 256, 128, 3 * 256 * 128);

    // ---- degrees, norm scales ----
    k_count<<<blocks(nEdgeRows), BT, 0, stream>>>(esrc, edst, cntOut, cntIn, nEdgeRows, E, N);
    k_scales<<<blocks(TOT), BT, 0, stream>>>(cntOut, sOut, TOT);
    k_scales<<<blocks(TOT), BT, 0, stream>>>(cntIn, sIn, TOT);

    // ---- build dst-grouped CSR (atomic-free SpMM afterwards) ----
    k_scan1<<<nChunks, 256, 0, stream>>>(cntIn, offs, chunkT, TOT);
    k_scan2<<<1, 512, 0, stream>>>(chunkT, nChunks);
    k_scan3<<<blocks(TOT), BT, 0, stream>>>(offs, chunkT, TOT);
    k_copy_i32<<<blocks(TOT), BT, 0, stream>>>(offs, cursor, TOT);
    k_fill<<<blocks(nEdgeRows), BT, 0, stream>>>(edst, cursor, eidx, nEdgeRows, E, N);

    const int mTiles = N / 16;  // 100000 % 16 == 0

    // ---- layer 1: per relation (scale->bf16, WMMA GEMM, CSR gather) ----
    for (int r = 0; r < 3; ++r) {
        k_scale_to_bf16<<<blocks((long long)N * 128), BT, 0, stream>>>(
            x, sOut + (size_t)r * N, Abf, 128, (long long)N * 128);
        k_gemm_wmma<128, 256, 4><<<mTiles, 128, 0, stream>>>(
            Abf, W1T + (size_t)r * 256 * 128, tmp);
        k_gather<256><<<blocks((long long)N * 64), BT, 0, stream>>>(
            tmp, esrc, eidx, offs, cntIn, sIn + (size_t)r * N, acc1, N, r * N);
    }
    k_finalize<256, true><<<blocks((long long)N * 256), BT, 0, stream>>>(
        acc1, b1, (long long)N * 256);

    // ---- layer 2 ----
    for (int r = 0; r < 3; ++r) {
        k_scale_to_bf16<<<blocks((long long)N * 256), BT, 0, stream>>>(
            acc1, sOut + (size_t)r * N, Abf, 256, (long long)N * 256);
        k_gemm_wmma<256, 128, 2><<<mTiles, 64, 0, stream>>>(
            Abf, W2T + (size_t)r * 128 * 256, tmp);
        k_gather<128><<<blocks((long long)N * 32), BT, 0, stream>>>(
            tmp, esrc, eidx, offs, cntIn, sIn + (size_t)r * N, acc2, N, r * N);
    }
    k_finalize<128, false><<<blocks((long long)N * 128), BT, 0, stream>>>(
        acc2, b2, (long long)N * 128);

    // ---- gather + relu + dot(linW) + sigmoid over 1.4M rows ----
    k_score<<<blocks((long long)nTotal * 32), BT, 0, stream>>>(
        acc2, esrc, edst, pairs, linW, linb, out, nEdgeRows, nTotal);
}
